// ViT_76802605187280
// MI455X (gfx1250) — matrix operand order repaired and checked by reading, not compiled, not run on hardware
//
#include <hip/hip_runtime.h>
#include <hip/hip_bf16.h>
#include <math.h>

// ---------------- problem constants ----------------
#define N_    64
#define D_    768
#define H_    12
#define DH_   64
#define S_    197      // tokens incl. cls
#define SP_   224      // padded token count (multiple of 32 for WMMA K-loops)
#define TOK_  196      // patch tokens
#define IN_   768
#define O_    1000

// ---------------- WMMA types / helpers ----------------
typedef __attribute__((ext_vector_type(16))) __bf16          v16bf;
typedef __attribute__((ext_vector_type(16))) unsigned short  v16u;
typedef __attribute__((ext_vector_type(8)))  float           v8f;
typedef __attribute__((ext_vector_type(4)))  int             v4i;

// CDNA5 async global->LDS DMA (ASYNCcnt-tracked), guarded for toolchain support.
#if __has_builtin(__builtin_amdgcn_global_load_async_to_lds_b128)
#define HAVE_ASYNC_LDS 1
typedef __attribute__((address_space(1))) void gas_void;
typedef __attribute__((address_space(3))) void las_void;
typedef __attribute__((address_space(1))) v4i  gas_v4i;
typedef __attribute__((address_space(3))) v4i  las_v4i;

// issue one 16B async DMA global->LDS (signature: int4 AS1*, int4 AS3*, imm, imm)
__device__ __forceinline__ void async_copy16(const void* g, void* l) {
  gas_void* gp = (gas_void*)(void*)g;     // const drop + addrspacecast to global
  las_void* lp = (las_void*)l;            // addrspacecast to LDS
  __builtin_amdgcn_global_load_async_to_lds_b128((gas_v4i*)gp, (las_v4i*)lp, 0, 0);
}
#endif

__device__ __forceinline__ unsigned short f2bf(float f) {
  unsigned u = __builtin_bit_cast(unsigned, f);
  u += 0x7FFFu + ((u >> 16) & 1u);              // round-to-nearest-even
  return (unsigned short)(u >> 16);
}

__device__ __forceinline__ v8f wmma_bf16(v16bf a, v16bf b, v8f c) {
  // (neg_a, A, neg_b, B, c_mod, C, reuse_a, reuse_b)
  return __builtin_amdgcn_wmma_f32_16x16x32_bf16(false, a, false, b, (short)0, c,
                                                 false, false);
}

// A fragment: 16x32 tile of a row-major bf16 matrix; `A` points at tile origin.
// Lane L holds row L%16; K split at 16: lanes<16 -> k 0..15, lanes>=16 -> k 16..31.
__device__ __forceinline__ v16bf load_a_frag(const unsigned short* A, int lda, int lane) {
  const unsigned short* p = A + (lane & 15) * lda + ((lane >> 4) * 16);
  v16u t;
#pragma unroll
  for (int i = 0; i < 16; ++i) t[i] = p[i];
  return __builtin_bit_cast(v16bf, t);
}

// B fragment for C = A @ W^T, W row-major (Ncols x K); `W` points at tile origin
// (row nBase, col kBase). Lane L holds output column L%16; K split at 16.
__device__ __forceinline__ v16bf load_bT_frag(const unsigned short* W, int ldw, int lane) {
  const unsigned short* p = W + (lane & 15) * ldw + ((lane >> 4) * 16);
  v16u t;
#pragma unroll
  for (int i = 0; i < 16; ++i) t[i] = p[i];
  return __builtin_bit_cast(v16bf, t);
}

// B fragment for C = A @ B, B row-major (K x Ncols); `B` points at tile origin.
__device__ __forceinline__ v16bf load_bN_frag(const unsigned short* B, int ldb, int lane) {
  const unsigned short* p = B + ((lane >> 4) * 16) * ldb + (lane & 15);
  v16u t;
#pragma unroll
  for (int i = 0; i < 16; ++i) t[i] = p[i * ldb];
  return __builtin_bit_cast(v16bf, t);
}

// ---------------- kernels ----------------

__global__ void k_cvt(const float* __restrict__ src, unsigned short* __restrict__ dst, int n) {
  int i = blockIdx.x * blockDim.x + threadIdx.x;
  int stride = gridDim.x * blockDim.x;
  for (; i < n; i += stride) dst[i] = f2bf(src[i]);
}

// emb[r,:] = x_flat[r,:] @ W_map^T + b_map   (M=12544, N=768, K=768), flat output
// (no row->token index math here; the gather happens in k_cls_pos)
__global__ __launch_bounds__(128) void k_patch_embed(
    const unsigned short* __restrict__ xb, const unsigned short* __restrict__ Wb,
    const float* __restrict__ bias, float* __restrict__ emb) {
  int lane = threadIdx.x, wave = threadIdx.y;
  int mBase = blockIdx.x * 16;
  int nBase = (blockIdx.y * 4 + wave) * 16;
  v8f c = {};
  for (int k = 0; k < IN_; k += 32) {
    v16bf a = load_a_frag(xb + mBase * IN_ + k, IN_, lane);
    v16bf b = load_bT_frag(Wb + nBase * IN_ + k, IN_, lane);
    c = wmma_bf16(a, b, c);
  }
  int col = nBase + (lane & 15);
  float bv = bias[col];
  int m0 = mBase + (lane >> 4) * 8;
  float* outp = emb + (size_t)m0 * D_ + col;
#pragma unroll
  for (int v = 0; v < 8; ++v) outp[(size_t)v * D_] = c[v] + bv;
}

// gather flat embeddings into padded token layout, add cls row + sinusoidal
// pos-emb, zero the padding rows 197..223
__global__ void k_cls_pos(const float* __restrict__ emb, const float* __restrict__ cls,
                          float* __restrict__ tok) {
  int n = blockIdx.y, s = blockIdx.x;
  float* row = tok + ((size_t)n * SP_ + s) * D_;
  const float* src = emb + ((size_t)n * TOK_ + (s - 1)) * D_;   // valid for s>=1
  for (int d = threadIdx.x; d < D_; d += blockDim.x) {
    if (s >= S_) { row[d] = 0.0f; continue; }
    float expo = (float)(d & ~1) / (float)D_;
    float arg = (float)s * __powf(10000.0f, -expo);
    float pe = (d & 1) ? __cosf(arg) : __sinf(arg);
    float base = (s == 0) ? cls[d] : src[d];
    row[d] = base + pe;
  }
}

// LayerNorm over the whole (S,D) slab per sample; writes f32 + bf16, zeros pad rows
__global__ void k_ln(const float* __restrict__ xin, const float* __restrict__ g,
                     const float* __restrict__ b, float* __restrict__ yf,
                     unsigned short* __restrict__ yb) {
  int n = blockIdx.x;
  const float* X = xin + (size_t)n * SP_ * D_;
  __shared__ float s1[256], s2[256];
  float a0 = 0.f, a1 = 0.f;
  for (int i = threadIdx.x; i < S_ * D_; i += 256) { float v = X[i]; a0 += v; a1 += v * v; }
  s1[threadIdx.x] = a0; s2[threadIdx.x] = a1;
  __syncthreads();
  for (int o = 128; o > 0; o >>= 1) {
    if (threadIdx.x < o) { s1[threadIdx.x] += s1[threadIdx.x + o]; s2[threadIdx.x] += s2[threadIdx.x + o]; }
    __syncthreads();
  }
  float mean = s1[0] / (float)(S_ * D_);
  float var  = s2[0] / (float)(S_ * D_) - mean * mean;
  float inv  = rsqrtf(var + 1e-5f);
  size_t base = (size_t)n * SP_ * D_;
  for (int i = threadIdx.x; i < SP_ * D_; i += 256) {
    if (i < S_ * D_) {
      float v = (X[i] - mean) * inv * g[i] + b[i];
      yf[base + i] = v;
      yb[base + i] = f2bf(v);
    } else {
      yf[base + i] = 0.0f;
      yb[base + i] = 0;
    }
  }
}

// q/k/v = t_h @ W{q,k,v}[h]^T + bias  (per (n,h); q pre-scaled by 1/8)
__global__ __launch_bounds__(128) void k_qkv(
    const unsigned short* __restrict__ tbf, const unsigned short* __restrict__ Wqkv,
    const float* __restrict__ bq, const float* __restrict__ bk, const float* __restrict__ bvv,
    unsigned short* __restrict__ qb, unsigned short* __restrict__ kb,
    unsigned short* __restrict__ vb) {
  int lane = threadIdx.x, wave = threadIdx.y;
  int mBase = blockIdx.x * 16;
  int nh = blockIdx.y;
  int n = nh / H_, h = nh - n * H_;
  int z = blockIdx.z;
  const unsigned short* A = tbf + (size_t)n * SP_ * D_ + h * DH_;
  const unsigned short* W = Wqkv + ((size_t)(z * H_ + h)) * DH_ * DH_ + wave * 16 * DH_;
  v8f c = {};
  for (int k = 0; k < DH_; k += 32) {
    v16bf a = load_a_frag(A + (size_t)mBase * D_ + k, D_, lane);
    v16bf b = load_bT_frag(W + k, DH_, lane);
    c = wmma_bf16(a, b, c);
  }
  const float* bias = (z == 0) ? bq : (z == 1) ? bk : bvv;
  int e = wave * 16 + (lane & 15);
  float bval = bias[h * DH_ + e];
  float scale = (z == 0) ? 0.125f : 1.0f;   // fold 1/sqrt(DH) into q
  unsigned short* out = ((z == 0) ? qb : (z == 1) ? kb : vb) + (size_t)nh * SP_ * DH_;
  int m0 = mBase + (lane >> 4) * 8;
#pragma unroll
  for (int v = 0; v < 8; ++v)
    out[(size_t)(m0 + v) * DH_ + e] = f2bf((c[v] + bval) * scale);
}

// fused attention: issue async V->LDS DMA, compute scores via WMMA while DMA is
// in flight, masked softmax, probs@V via WMMA, residual add.
__global__ __launch_bounds__(128) void k_attn(
    const unsigned short* __restrict__ qb, const unsigned short* __restrict__ kb,
    const unsigned short* __restrict__ vb, const float* __restrict__ tokln,
    float* __restrict__ res) {
  __shared__ unsigned short vlds[SP_ * DH_];   // 28 KB
  __shared__ float          sc[16 * SP_];      // 14 KB
  __shared__ unsigned short pr[16 * SP_];      //  7 KB
  int lane = threadIdx.x, wave = threadIdx.y;
  int tid = wave * 32 + lane;
  int mBase = blockIdx.x * 16;                 // q-row tile (0..12 covers 197 rows)
  int nh = blockIdx.y;
  int n = nh / H_, h = nh - n * H_;
  const unsigned short* Q = qb + (size_t)nh * SP_ * DH_;
  const unsigned short* K = kb + (size_t)nh * SP_ * DH_;
  const unsigned short* V = vb + (size_t)nh * SP_ * DH_;

  // ---- stage V into LDS ----
#if HAVE_ASYNC_LDS
  // 16B per op; 1792 ops total; 14 per thread. DMA overlaps the scores GEMM.
#pragma unroll
  for (int i = 0; i < (SP_ * DH_ / 8) / 128; ++i) {
    int e = (i * 128 + tid) * 8;
    async_copy16(V + e, vlds + e);
  }
#else
  for (int i = tid; i < SP_ * DH_ / 4; i += 128)
    ((unsigned long long*)vlds)[i] = ((const unsigned long long*)V)[i];
#endif

  // ---- scores = q @ k^T (q already scaled by 1/8) ----
  for (int ct = wave; ct < SP_ / 16; ct += 4) {
    v8f c = {};
    for (int k = 0; k < DH_; k += 32) {
      v16bf a = load_a_frag(Q + (size_t)mBase * DH_ + k, DH_, lane);
      v16bf b = load_bT_frag(K + (size_t)ct * 16 * DH_ + k, DH_, lane);
      c = wmma_bf16(a, b, c);
    }
    int col = ct * 16 + (lane & 15);
    int r0 = (lane >> 4) * 8;
#pragma unroll
    for (int v = 0; v < 8; ++v) sc[(r0 + v) * SP_ + col] = c[v];
  }

#if HAVE_ASYNC_LDS
#if __has_builtin(__builtin_amdgcn_s_wait_asynccnt)
  __builtin_amdgcn_s_wait_asynccnt(0);
#else
  asm volatile("s_wait_asynccnt 0x0" ::: "memory");
#endif
#endif
  __syncthreads();

  // ---- masked softmax over 197 keys; 8 lanes per row, 28 cols per lane ----
  {
    int row = tid >> 3, sub = tid & 7;
    float mx = -3.4e38f;
#pragma unroll
    for (int i = 0; i < 28; ++i) {
      int cx = sub * 28 + i;
      float v = (cx < S_) ? sc[row * SP_ + cx] : -3.4e38f;
      mx = fmaxf(mx, v);
    }
    mx = fmaxf(mx, __shfl_xor(mx, 1));
    mx = fmaxf(mx, __shfl_xor(mx, 2));
    mx = fmaxf(mx, __shfl_xor(mx, 4));
    float pv[28];
    float sum = 0.f;
#pragma unroll
    for (int i = 0; i < 28; ++i) {
      int cx = sub * 28 + i;
      float v = (cx < S_) ? __expf(sc[row * SP_ + cx] - mx) : 0.0f;
      pv[i] = v; sum += v;
    }
    sum += __shfl_xor(sum, 1);
    sum += __shfl_xor(sum, 2);
    sum += __shfl_xor(sum, 4);
    float rs = 1.0f / sum;
#pragma unroll
    for (int i = 0; i < 28; ++i) {
      int cx = sub * 28 + i;
      pr[row * SP_ + cx] = f2bf(pv[i] * rs);   // pad cols -> exact 0
    }
  }
  __syncthreads();

  // ---- msa = probs @ V ; write residual tokens + msa ----
  v8f c = {};
  for (int k = 0; k < SP_; k += 32) {
    v16bf a = load_a_frag(pr + k, SP_, lane);
    v16bf b = load_bN_frag(vlds + (size_t)k * DH_ + wave * 16, DH_, lane);
    c = wmma_bf16(a, b, c);
  }
  int col = h * DH_ + wave * 16 + (lane & 15);
  int r0 = mBase + (lane >> 4) * 8;
#pragma unroll
  for (int v = 0; v < 8; ++v) {
    size_t idx = ((size_t)n * SP_ + r0 + v) * D_ + col;
    res[idx] = tokln[idx] + c[v];
  }
}

// LN2: full (S,D) moments, but only row 0 is ever consumed downstream
__global__ void k_ln_row0(const float* __restrict__ xin, const float* __restrict__ g,
                          const float* __restrict__ b, unsigned short* __restrict__ y) {
  int n = blockIdx.x;
  const float* X = xin + (size_t)n * SP_ * D_;
  __shared__ float s1[256], s2[256];
  float a0 = 0.f, a1 = 0.f;
  for (int i = threadIdx.x; i < S_ * D_; i += 256) { float v = X[i]; a0 += v; a1 += v * v; }
  s1[threadIdx.x] = a0; s2[threadIdx.x] = a1;
  __syncthreads();
  for (int o = 128; o > 0; o >>= 1) {
    if (threadIdx.x < o) { s1[threadIdx.x] += s1[threadIdx.x + o]; s2[threadIdx.x] += s2[threadIdx.x + o]; }
    __syncthreads();
  }
  float mean = s1[0] / (float)(S_ * D_);
  float var  = s2[0] / (float)(S_ * D_) - mean * mean;
  float inv  = rsqrtf(var + 1e-5f);
  for (int d = threadIdx.x; d < D_; d += 256)
    y[(size_t)n * D_ + d] = f2bf((X[d] - mean) * inv * g[d] + b[d]);
}

// mlp_row0 = gelu(A @ W_mlp^T + b)  (M=64, N=768, K=768)
__global__ __launch_bounds__(128) void k_mlp(
    const unsigned short* __restrict__ A, const unsigned short* __restrict__ Wb,
    const float* __restrict__ bias, unsigned short* __restrict__ outb) {
  int lane = threadIdx.x, wave = threadIdx.y;
  int mBase = blockIdx.x * 16;
  int nBase = (blockIdx.y * 4 + wave) * 16;
  v8f c = {};
  for (int k = 0; k < D_; k += 32) {
    v16bf a = load_a_frag(A + (size_t)mBase * D_ + k, D_, lane);
    v16bf b = load_bT_frag(Wb + (size_t)nBase * D_ + k, D_, lane);
    c = wmma_bf16(a, b, c);
  }
  int col = nBase + (lane & 15);
  float bv = bias[col];
  int m0 = mBase + (lane >> 4) * 8;
#pragma unroll
  for (int v = 0; v < 8; ++v) {
    float x = c[v] + bv;
    float g = 0.5f * x * (1.0f + erff(x * 0.70710678f));   // exact gelu
    outb[(size_t)(m0 + v) * D_ + col] = f2bf(g);
  }
}

// logits = A @ W_head^T + b_head  (M=64, N=1000 padded to 1008, K=768)
__global__ __launch_bounds__(128) void k_head(
    const unsigned short* __restrict__ A, const unsigned short* __restrict__ Wb,
    const float* __restrict__ bias, float* __restrict__ logits) {
  int lane = threadIdx.x, wave = threadIdx.y;
  int nt = blockIdx.y * 4 + wave;
  if (nt > 62) return;                          // tile 62 covers cols 992..1007
  int mBase = blockIdx.x * 16;
  int nBase = nt * 16;
  int nrow = nBase + (lane & 15);
  if (nrow > O_ - 1) nrow = O_ - 1;             // clamp OOB weight rows
  v8f c = {};
  for (int k = 0; k < D_; k += 32) {
    v16bf a = load_a_frag(A + (size_t)mBase * D_ + k, D_, lane);
    const unsigned short* p = Wb + (size_t)nrow * D_ + k + ((lane >> 4) * 16);
    v16u t;
#pragma unroll
    for (int i = 0; i < 16; ++i) t[i] = p[i];
    v16bf b = __builtin_bit_cast(v16bf, t);
    c = wmma_bf16(a, b, c);
  }
  int col = nBase + (lane & 15);
  int m0 = mBase + (lane >> 4) * 8;
  if (col < O_) {
    float bv = bias[col];
#pragma unroll
    for (int v = 0; v < 8; ++v) logits[(size_t)(m0 + v) * O_ + col] = c[v] + bv;
  }
}

__global__ void k_softmax(const float* __restrict__ logits, float* __restrict__ out) {
  int n = blockIdx.x;
  const float* L = logits + (size_t)n * O_;
  __shared__ float sm[256];
  float mx = -3.4e38f;
  for (int i = threadIdx.x; i < O_; i += 256) mx = fmaxf(mx, L[i]);
  sm[threadIdx.x] = mx; __syncthreads();
  for (int o = 128; o > 0; o >>= 1) {
    if (threadIdx.x < o) sm[threadIdx.x] = fmaxf(sm[threadIdx.x], sm[threadIdx.x + o]);
    __syncthreads();
  }
  mx = sm[0]; __syncthreads();
  float s = 0.f;
  for (int i = threadIdx.x; i < O_; i += 256) s += __expf(L[i] - mx);
  sm[threadIdx.x] = s; __syncthreads();
  for (int o = 128; o > 0; o >>= 1) {
    if (threadIdx.x < o) sm[threadIdx.x] += sm[threadIdx.x + o];
    __syncthreads();
  }
  float rs = 1.0f / sm[0];
  for (int i = threadIdx.x; i < O_; i += 256) out[(size_t)n * O_ + i] = __expf(L[i] - mx) * rs;
}

// ---------------- launch ----------------
extern "C" void kernel_launch(void* const* d_in, const int* in_sizes, int n_in,
                              void* d_out, int out_size, void* d_ws, size_t ws_size,
                              hipStream_t stream) {
  (void)in_sizes; (void)n_in; (void)out_size; (void)ws_size;
  const float* x      = (const float*)d_in[0];
  const float* W_map  = (const float*)d_in[1];
  const float* b_map  = (const float*)d_in[2];
  const float* cls    = (const float*)d_in[3];
  const float* g1     = (const float*)d_in[4];
  const float* b1     = (const float*)d_in[5];
  const float* Wq     = (const float*)d_in[6];
  const float* bq     = (const float*)d_in[7];
  const float* Wk     = (const float*)d_in[8];
  const float* bk     = (const float*)d_in[9];
  const float* Wv     = (const float*)d_in[10];
  const float* bv     = (const float*)d_in[11];
  const float* g2     = (const float*)d_in[12];
  const float* b2     = (const float*)d_in[13];
  const float* W_mlp  = (const float*)d_in[14];
  const float* b_mlp  = (const float*)d_in[15];
  const float* W_head = (const float*)d_in[16];
  const float* b_head = (const float*)d_in[17];
  float* out = (float*)d_out;

  // workspace carve-out (~285 MB)
  char* p = (char*)d_ws;
  auto carve = [&](size_t bytes) {
    char* r = p;
    p += (bytes + 255) & ~(size_t)255;
    return (void*)r;
  };
  const size_t nTok = (size_t)N_ * SP_ * D_;                 // 11,010,048
  unsigned short* xb     = (unsigned short*)carve((size_t)N_ * TOK_ * IN_ * 2);
  unsigned short* Wmapb  = (unsigned short*)carve((size_t)D_ * IN_ * 2);
  unsigned short* Wqkvb  = (unsigned short*)carve((size_t)3 * H_ * DH_ * DH_ * 2);
  unsigned short* Wmlpb  = (unsigned short*)carve((size_t)D_ * D_ * 2);
  unsigned short* Wheadb = (unsigned short*)carve((size_t)O_ * D_ * 2);
  float*          emb    = (float*)carve((size_t)N_ * TOK_ * D_ * 4);  // flat GEMM out
  float*          tok    = (float*)carve(nTok * 4);
  float*          tokln  = (float*)carve(nTok * 4);
  unsigned short* tokbf  = (unsigned short*)carve(nTok * 2);
  unsigned short* qb     = (unsigned short*)carve(nTok * 2); // (n*H+h, SP, DH)
  unsigned short* kb     = (unsigned short*)carve(nTok * 2);
  unsigned short* vb     = (unsigned short*)carve(nTok * 2);
  float*          res    = (float*)carve(nTok * 4);
  unsigned short* row0   = (unsigned short*)carve((size_t)N_ * D_ * 2);
  unsigned short* mlpo   = (unsigned short*)carve((size_t)N_ * D_ * 2);
  float*          logits = (float*)carve((size_t)N_ * O_ * 4);

  auto nb = [](int n) { return (n + 255) / 256; };
  const int QKVW = H_ * DH_ * DH_;   // 49152 per matrix

  // fp32 -> bf16 conversions
  k_cvt<<<1024, 256, 0, stream>>>(x, xb, N_ * TOK_ * IN_);
  k_cvt<<<nb(D_ * IN_), 256, 0, stream>>>(W_map, Wmapb, D_ * IN_);
  k_cvt<<<nb(QKVW), 256, 0, stream>>>(Wq, Wqkvb + 0 * QKVW, QKVW);
  k_cvt<<<nb(QKVW), 256, 0, stream>>>(Wk, Wqkvb + 1 * QKVW, QKVW);
  k_cvt<<<nb(QKVW), 256, 0, stream>>>(Wv, Wqkvb + 2 * QKVW, QKVW);
  k_cvt<<<nb(D_ * D_), 256, 0, stream>>>(W_mlp, Wmlpb, D_ * D_);
  k_cvt<<<nb(O_ * D_), 256, 0, stream>>>(W_head, Wheadb, O_ * D_);

  // patch embedding GEMM (12544x768x768) -> flat emb
  k_patch_embed<<<dim3(784, 12), dim3(32, 4), 0, stream>>>(xb, Wmapb, b_map, emb);
  // gather + cls token + positional embedding + zero padding rows
  k_cls_pos<<<dim3(SP_, N_), 256, 0, stream>>>(emb, cls, tok);
  // LayerNorm 1 over (S,D) per sample
  k_ln<<<N_, 256, 0, stream>>>(tok, g1, b1, tokln, tokbf);
  // QKV projections (per head 64x64), q scaled by 1/8
  k_qkv<<<dim3(SP_ / 16, N_ * H_, 3), dim3(32, 4), 0, stream>>>(tokbf, Wqkvb, bq, bk, bv,
                                                                qb, kb, vb);
  // fused attention + residual (13 q-row tiles cover 197 rows)
  k_attn<<<dim3(13, N_ * H_), dim3(32, 4), 0, stream>>>(qb, kb, vb, tokln, res);
  // LayerNorm 2; only row 0 feeds the classifier path
  k_ln_row0<<<N_, 256, 0, stream>>>(res, g2, b2, row0);
  // MLP (row 0 only) with exact GELU
  k_mlp<<<dim3(4, 12), dim3(32, 4), 0, stream>>>(row0, Wmlpb, b_mlp, mlpo);
  // classifier head (1000 cols, padded tiles with clamped loads)
  k_head<<<dim3(4, 16), dim3(32, 4), 0, stream>>>(mlpo, Wheadb, b_head, logits);
  // final softmax -> d_out
  k_softmax<<<N_, 256, 0, stream>>>(logits, out);
}
